// GConvEn_64828236365869
// MI455X (gfx1250) — matrix operand order repaired and checked by reading, not compiled, not run on hardware
//
#include <hip/hip_runtime.h>
#include <hip/hip_bf16.h>
#include <cstdint>
#include <cstddef>

// ---------------- problem constants ----------------
constexpr int Bsz = 2, Nn = 4096, Dd = 128, KNN = 32, CF = 16;
constexpr int EI  = 2 * Dd + 1;          // 257
constexpr int H1  = 2 * EI;              // 514
// edge GEMM1: [64 x 288] @ [288 x 544]
constexpr int K1PAD = 288;  constexpr int KS1 = K1PAD / 32;   // 9
constexpr int N1PAD = 544;  constexpr int NT1 = N1PAD / 16;   // 34
// edge GEMM2: [64 x 544] @ [544 x 16]
constexpr int KS2 = N1PAD / 32;                               // 17
// node GEMM1: [rows x 160] @ [160 x 256]
constexpr int NK1 = Dd + CF;                                  // 144
constexpr int NK1PAD = 160; constexpr int NKS1 = 5;
constexpr int NH = 2 * Dd;                                    // 256
constexpr int NNT1 = NH / 16;                                 // 16
// node GEMM2: [rows x 256] @ [256 x 128]
constexpr int NKS2 = NH / 32;                                 // 8
constexpr int NNT2 = Dd / 16;                                 // 8
// knn split
constexpr int SPLIT = 4;
constexpr int PART  = Nn / SPLIT;                             // 1024

typedef __bf16 v16bf __attribute__((ext_vector_type(16)));
typedef float  v8f   __attribute__((ext_vector_type(8)));

union FragBF { uint4 q[2]; v16bf v; };

static __device__ __forceinline__ unsigned short f2bf(float f) {
  unsigned int u = __float_as_uint(f);
  unsigned int r = u + 0x7FFFu + ((u >> 16) & 1u);   // round-to-nearest-even
  return (unsigned short)(r >> 16);
}
static __device__ __forceinline__ float silu_f(float x) {
  return x / (1.0f + __expf(-x));
}

// ---------------- weight pre-pack into WMMA B-fragment layout ----------------
// lane L: N = ntile*16 + (L&15); K values = kstep*32 + (L<16?0:8) + {[0..7], 16+[0..7]}
__global__ __launch_bounds__(256) void prepack_kernel(
    const float* __restrict__ W, unsigned short* __restrict__ P,
    int Kdim, int Ndim, int ksteps, int ntiles)
{
  int frag  = blockIdx.x;                 // kstep * ntiles + ntile
  int kstep = frag / ntiles, ntile = frag % ntiles;
  for (int e = threadIdx.x; e < 512; e += 256) {
    int lane = e >> 4;
    int kidx = e & 15;
    int kk = kstep * 32 + ((lane & 16) ? 8 : 0) + (kidx & 7) + ((kidx & 8) ? 16 : 0);
    int nn = ntile * 16 + (lane & 15);
    float val = (kk < Kdim && nn < Ndim) ? W[(size_t)kk * Ndim + nn] : 0.0f;
    P[(size_t)frag * 512 + e] = f2bf(val);
  }
}

// ---------------- KNN phase 1: each thread scans N/SPLIT candidates ----------------
__global__ __launch_bounds__(256) void knn_scan_kernel(
    const float* __restrict__ x, unsigned long long* __restrict__ part_out)
{
  int flat = blockIdx.x * 256 + threadIdx.x;   // over Nn * SPLIT
  int b = blockIdx.y;
  int i = flat >> 2;
  int part = flat & 3;
  const float* xb = x + (size_t)b * Nn * 3;
  float xi0 = xb[i * 3 + 0], xi1 = xb[i * 3 + 1], xi2 = xb[i * 3 + 2];
  float bd[KNN]; int bi[KNN];
#pragma unroll
  for (int k = 0; k < KNN; ++k) { bd[k] = 3.4e38f; bi[k] = 0; }
  int j0 = part * PART;
  for (int jj = 0; jj < PART; ++jj) {
    int j = j0 + jj;
    float d0 = xi0 - xb[j * 3 + 0];
    float d1 = xi1 - xb[j * 3 + 1];
    float d2 = xi2 - xb[j * 3 + 2];
    float d = d0 * d0 + d1 * d1 + d2 * d2;
    if (d < bd[KNN - 1]) {
      bd[KNN - 1] = d; bi[KNN - 1] = j;
#pragma unroll
      for (int k = KNN - 2; k >= 0; --k) {      // one bubble pass = insertion
        if (bd[k + 1] < bd[k]) {
          float td = bd[k]; bd[k] = bd[k + 1]; bd[k + 1] = td;
          int   ti = bi[k]; bi[k] = bi[k + 1]; bi[k + 1] = ti;
        }
      }
    }
  }
  size_t base = (((size_t)b * Nn + i) * SPLIT + part) * KNN;
#pragma unroll
  for (int k = 0; k < KNN; ++k)
    part_out[base + k] =
        ((unsigned long long)__float_as_uint(bd[k]) << 32) | (unsigned int)bi[k];
}

// ---------------- KNN phase 2: exact 4-way merge of sorted runs ----------------
__global__ __launch_bounds__(256) void knn_merge_kernel(
    const unsigned long long* __restrict__ part_in,
    int* __restrict__ idxb, float* __restrict__ distb)
{
  int q = blockIdx.x * 256 + threadIdx.x;      // 0 .. B*N-1
  size_t base = (size_t)q * SPLIT * KNN;
  int cur[SPLIT];
  unsigned long long head[SPLIT];
#pragma unroll
  for (int p = 0; p < SPLIT; ++p) { cur[p] = 0; head[p] = part_in[base + p * KNN]; }
  for (int k = 0; k < KNN; ++k) {
    unsigned long long best = head[0]; int bp = 0;
#pragma unroll
    for (int p = 1; p < SPLIT; ++p)
      if (head[p] < best) { best = head[p]; bp = p; }
    cur[bp]++;
    head[bp] = (cur[bp] < KNN) ? part_in[base + bp * KNN + cur[bp]] : ~0ull;
    idxb[(size_t)q * KNN + k]  = (int)(unsigned int)best;
    distb[(size_t)q * KNN + k] = __uint_as_float((unsigned int)(best >> 32));
  }
}

// ---------------- fused edge kernel: TWO nodes per block (64 edge rows) ----------------
__global__ __launch_bounds__(256) void edge_kernel(
    const float* __restrict__ h, const float* __restrict__ x,
    const unsigned char* __restrict__ mask,
    const float* __restrict__ be1, const float* __restrict__ be2,
    const float* __restrict__ Wc1, const float* __restrict__ bc1,
    const float* __restrict__ Wc2, const float* __restrict__ bc2,
    const unsigned short* __restrict__ pWe1, const unsigned short* __restrict__ pWe2,
    const int* __restrict__ idxb, const float* __restrict__ distb,
    float* __restrict__ m_i_out, float* __restrict__ coors_out)
{
  constexpr int ROWS = 64;
  constexpr int RSA = 296;   // row stride (ushort), 16B-aligned rows
  constexpr int RSH = 560;
  __shared__ unsigned short As[ROWS * RSA];   // 37,888 B
  __shared__ unsigned short Hs[ROWS * RSH];   // 71,680 B
  __shared__ float m_s[ROWS][CF];             // 4 KB
  __shared__ float cw_s[ROWS];
  __shared__ int   idx_s[ROWS];
  __shared__ float dst_s[ROWS];

  int g  = blockIdx.x;          // 0 .. B*N/2-1
  int n0 = g * 2;               // flat node index of first node (never crosses batch)
  int b  = n0 / Nn;
  int i0 = n0 % Nn;
  int t = threadIdx.x, lane = t & 31, wave = t >> 5;

  if (t < ROWS) {
    size_t eidx = (size_t)(n0 + (t >> 5)) * KNN + (t & 31);
    idx_s[t] = idxb[eidx];
    dst_s[t] = distb[eidx];
  }
  __syncthreads();

  // build A: row r -> node (r/32), neighbor (r%32); cols: h_i | h_j | dist | pad
  const float* hb = h + (size_t)b * Nn * Dd;
  for (int e = t; e < ROWS * K1PAD; e += 256) {
    int row = e / K1PAD, col = e % K1PAD;
    int irow = i0 + (row >> 5);
    float v;
    if (col < Dd)            v = hb[(size_t)irow * Dd + col];
    else if (col < 2 * Dd)   v = hb[(size_t)idx_s[row] * Dd + (col - Dd)];
    else if (col == 2 * Dd)  v = dst_s[row];
    else                     v = 0.0f;
    As[row * RSA + col] = f2bf(v);
  }
  __syncthreads();

  int m   = lane & 15;
  int kb  = (lane & 16) ? 8 : 0;
  int hi8 = (lane & 16) ? 8 : 0;

  // GEMM1: [64 x 288] @ [288 x 544]; B fragments hoisted to regs, reused by 4 M-tiles
  for (int nt = wave; nt < NT1; nt += 8) {
    FragBF bfr[KS1];
#pragma unroll
    for (int ks = 0; ks < KS1; ++ks) {
      const uint4* bp = (const uint4*)(pWe1 + (size_t)(ks * NT1 + nt) * 512) + lane * 2;
      bfr[ks].q[0] = bp[0]; bfr[ks].q[1] = bp[1];
    }
    int col = nt * 16 + m;
    float bias = (col < H1) ? be1[col] : 0.0f;
#pragma unroll
    for (int mt = 0; mt < 4; ++mt) {
      v8f acc = {};
#pragma unroll
      for (int ks = 0; ks < KS1; ++ks) {
        FragBF a;
        const uint4* ap = (const uint4*)&As[(mt * 16 + m) * RSA + ks * 32 + kb];
        a.q[0] = ap[0]; a.q[1] = ap[2];
        acc = __builtin_amdgcn_wmma_f32_16x16x32_bf16(false, a.v, false, bfr[ks].v,
                                                      (short)0, acc, false, false);
      }
#pragma unroll
      for (int r = 0; r < 8; ++r) {
        int row = mt * 16 + r + hi8;
        Hs[row * RSH + col] = f2bf(silu_f(acc[r] + bias));
      }
    }
  }
  __syncthreads();

  // GEMM2: [64 x 544] @ [544 x 16]; waves 0..3 each own one M-tile; B double-buffered
  if (wave < 4) {
    int mt = wave;
    v8f acc = {};
    FragBF bcur, bnxt;
    {
      const uint4* bp = (const uint4*)pWe2 + lane * 2;
      bcur.q[0] = bp[0]; bcur.q[1] = bp[1];
    }
#pragma unroll
    for (int ks = 0; ks < KS2; ++ks) {
      if (ks + 1 < KS2) {
        const uint4* bp = (const uint4*)(pWe2 + (size_t)(ks + 1) * 512) + lane * 2;
        bnxt.q[0] = bp[0]; bnxt.q[1] = bp[1];
      }
      FragBF a;
      const uint4* ap = (const uint4*)&Hs[(mt * 16 + m) * RSH + ks * 32 + kb];
      a.q[0] = ap[0]; a.q[1] = ap[2];
      acc = __builtin_amdgcn_wmma_f32_16x16x32_bf16(false, a.v, false, bcur.v,
                                                    (short)0, acc, false, false);
      bcur = bnxt;
    }
    float bias = be2[m];
#pragma unroll
    for (int r = 0; r < 8; ++r) {
      int row = mt * 16 + r + hi8;
      m_s[row][m] = silu_f(acc[r] + bias);
    }
  }
  __syncthreads();

  // coors MLP per edge (16 -> 64 -> 1) + mask
  if (t < ROWS) {
    float acc = bc2[0];
    for (int u = 0; u < 4 * CF; ++u) {
      float s = bc1[u];
#pragma unroll
      for (int c = 0; c < CF; ++c) s += m_s[t][c] * Wc1[c * 4 * CF + u];
      acc += silu_f(s) * Wc2[u];
    }
    int irow = i0 + (t >> 5);
    bool mij = (mask[(size_t)b * Nn + irow] != 0) &&
               (mask[(size_t)b * Nn + idx_s[t]] != 0);
    cw_s[t] = mij ? acc : 0.0f;
  }
  __syncthreads();

  // m_i = sum_k m_ij       (32 threads: 2 nodes x 16 features)
  if (t < 2 * CF) {
    int ln = t >> 4, cf = t & 15;
    float s = 0.0f;
#pragma unroll
    for (int k = 0; k < KNN; ++k) s += m_s[ln * 32 + k][cf];
    m_i_out[(size_t)(n0 + ln) * CF + cf] = s;
  } else if (t >= 64 && t < 70) {
    // coors_out = x_i + sum_k cw_k * (x_i - x_j)   (6 threads: 2 nodes x 3 comps)
    int ln = (t - 64) / 3, c = (t - 64) % 3;
    int i = i0 + ln;
    const float* xb = x + (size_t)b * Nn * 3;
    float xi = xb[(size_t)i * 3 + c];
    float s = xi;
    for (int k = 0; k < KNN; ++k)
      s += cw_s[ln * 32 + k] * (xi - xb[(size_t)idx_s[ln * 32 + k] * 3 + c]);
    coors_out[(size_t)(n0 + ln) * 3 + c] = s;
  }
}

// ---------------- node MLP: [h | m_i] -> 256 -> 128, + residual ----------------
__global__ __launch_bounds__(256) void node_kernel(
    const float* __restrict__ h, const float* __restrict__ m_i,
    const float* __restrict__ bn1, const float* __restrict__ bn2,
    const unsigned short* __restrict__ pWn1, const unsigned short* __restrict__ pWn2,
    float* __restrict__ out)
{
  constexpr int ROWS = 64;
  constexpr int RSA = 168;
  constexpr int RSH = 264;
  __shared__ unsigned short As[ROWS * RSA];   // 21,504 B
  __shared__ unsigned short Hs[ROWS * RSH];   // 33,792 B

  int t = threadIdx.x, lane = t & 31, wave = t >> 5;
  int row0 = blockIdx.x * ROWS;               // flat node index b*N+n

  for (int e = t; e < ROWS * NK1PAD; e += 256) {
    int row = e / NK1PAD, col = e % NK1PAD;
    int gr = row0 + row;
    float v;
    if (col < Dd)        v = h[(size_t)gr * Dd + col];
    else if (col < NK1)  v = m_i[(size_t)gr * CF + (col - Dd)];
    else                 v = 0.0f;
    As[row * RSA + col] = f2bf(v);
  }
  __syncthreads();

  int m   = lane & 15;
  int kb  = (lane & 16) ? 8 : 0;
  int hi8 = (lane & 16) ? 8 : 0;

  // GEMM1: [64 x 160] @ [160 x 256]; 5 B fragments hoisted, reused by 4 M-tiles
  for (int nt = wave; nt < NNT1; nt += 8) {
    FragBF bfr[NKS1];
#pragma unroll
    for (int ks = 0; ks < NKS1; ++ks) {
      const uint4* bp = (const uint4*)(pWn1 + (size_t)(ks * NNT1 + nt) * 512) + lane * 2;
      bfr[ks].q[0] = bp[0]; bfr[ks].q[1] = bp[1];
    }
    int col = nt * 16 + m;
    float bias = bn1[col];
#pragma unroll
    for (int mt = 0; mt < 4; ++mt) {
      v8f acc = {};
#pragma unroll
      for (int ks = 0; ks < NKS1; ++ks) {
        FragBF a;
        const uint4* ap = (const uint4*)&As[(mt * 16 + m) * RSA + ks * 32 + kb];
        a.q[0] = ap[0]; a.q[1] = ap[2];
        acc = __builtin_amdgcn_wmma_f32_16x16x32_bf16(false, a.v, false, bfr[ks].v,
                                                      (short)0, acc, false, false);
      }
#pragma unroll
      for (int r = 0; r < 8; ++r) {
        int row = mt * 16 + r + hi8;
        Hs[row * RSH + col] = f2bf(silu_f(acc[r] + bias));
      }
    }
  }
  __syncthreads();

  // GEMM2: [64 x 256] @ [256 x 128] + bn2 + h; 8 B fragments hoisted per n-tile
  {
    int nt = wave;                             // 8 waves, 8 n-tiles
    FragBF bfr[NKS2];
#pragma unroll
    for (int ks = 0; ks < NKS2; ++ks) {
      const uint4* bp = (const uint4*)(pWn2 + (size_t)(ks * NNT2 + nt) * 512) + lane * 2;
      bfr[ks].q[0] = bp[0]; bfr[ks].q[1] = bp[1];
    }
    int col = nt * 16 + m;
    float bias = bn2[col];
#pragma unroll
    for (int mt = 0; mt < 4; ++mt) {
      v8f acc = {};
#pragma unroll
      for (int ks = 0; ks < NKS2; ++ks) {
        FragBF a;
        const uint4* ap = (const uint4*)&Hs[(mt * 16 + m) * RSH + ks * 32 + kb];
        a.q[0] = ap[0]; a.q[1] = ap[2];
        acc = __builtin_amdgcn_wmma_f32_16x16x32_bf16(false, a.v, false, bfr[ks].v,
                                                      (short)0, acc, false, false);
      }
#pragma unroll
      for (int r = 0; r < 8; ++r) {
        int row = mt * 16 + r + hi8;
        int gr = row0 + row;
        out[(size_t)gr * Dd + col] = acc[r] + bias + h[(size_t)gr * Dd + col];
      }
    }
  }
}

// ---------------- launch ----------------
extern "C" void kernel_launch(void* const* d_in, const int* in_sizes, int n_in,
                              void* d_out, int out_size, void* d_ws, size_t ws_size,
                              hipStream_t stream) {
  (void)in_sizes; (void)n_in; (void)out_size; (void)ws_size;
  const float* h   = (const float*)d_in[0];
  const float* x   = (const float*)d_in[1];
  const unsigned char* mask = (const unsigned char*)d_in[2];
  const float* We1 = (const float*)d_in[3];  const float* be1 = (const float*)d_in[4];
  const float* We2 = (const float*)d_in[5];  const float* be2 = (const float*)d_in[6];
  const float* Wc1 = (const float*)d_in[7];  const float* bc1 = (const float*)d_in[8];
  const float* Wc2 = (const float*)d_in[9];  const float* bc2 = (const float*)d_in[10];
  const float* Wn1 = (const float*)d_in[11]; const float* bn1 = (const float*)d_in[12];
  const float* Wn2 = (const float*)d_in[13]; const float* bn2 = (const float*)d_in[14];

  float* node_out  = (float*)d_out;                        // [B,N,D]
  float* coors_out = node_out + (size_t)Bsz * Nn * Dd;     // [B,N,3]

  char* ws = (char*)d_ws;
  unsigned long long* knn_parts = (unsigned long long*)ws;
  ws += (size_t)Bsz * Nn * SPLIT * KNN * 8;                // 8 MB
  int*   idxb  = (int*)ws;             ws += (size_t)Bsz * Nn * KNN * 4;
  float* distb = (float*)ws;           ws += (size_t)Bsz * Nn * KNN * 4;
  float* m_i   = (float*)ws;           ws += (size_t)Bsz * Nn * CF * 4;
  unsigned short* pWe1 = (unsigned short*)ws; ws += (size_t)KS1 * NT1 * 512 * 2;
  unsigned short* pWe2 = (unsigned short*)ws; ws += (size_t)KS2 * 1 * 512 * 2;
  unsigned short* pWn1 = (unsigned short*)ws; ws += (size_t)NKS1 * NNT1 * 512 * 2;
  unsigned short* pWn2 = (unsigned short*)ws; ws += (size_t)NKS2 * NNT2 * 512 * 2;

  prepack_kernel<<<KS1 * NT1, 256, 0, stream>>>(We1, pWe1, EI, H1, KS1, NT1);
  prepack_kernel<<<KS2, 256, 0, stream>>>(We2, pWe2, H1, CF, KS2, 1);
  prepack_kernel<<<NKS1 * NNT1, 256, 0, stream>>>(Wn1, pWn1, NK1, NH, NKS1, NNT1);
  prepack_kernel<<<NKS2 * NNT2, 256, 0, stream>>>(Wn2, pWn2, NH, Dd, NKS2, NNT2);

  knn_scan_kernel<<<dim3((Nn * SPLIT) / 256, Bsz), 256, 0, stream>>>(x, knn_parts);
  knn_merge_kernel<<<(Bsz * Nn) / 256, 256, 0, stream>>>(knn_parts, idxb, distb);

  edge_kernel<<<(Bsz * Nn) / 2, 256, 0, stream>>>(h, x, mask, be1, be2, Wc1, bc1,
                                                  Wc2, bc2, pWe1, pWe2, idxb, distb,
                                                  m_i, coors_out);

  node_kernel<<<(Bsz * Nn) / 64, 256, 0, stream>>>(h, m_i, bn1, bn2, pWn1, pWn2, node_out);
}
